// FenwickTree_67070209294948
// MI455X (gfx1250) — compile-verified, matrix operand order checked by reading, not run on hardware
//
#include <hip/hip_runtime.h>

typedef __attribute__((ext_vector_type(16))) _Float16 v16h;
typedef __attribute__((ext_vector_type(2)))  _Float16 v2h;
typedef __attribute__((ext_vector_type(8)))  float    v8f;

#define D_    128
#define NG    640       // 5*D gate columns
#define M2_   65536
#define M_    131072
#define KTOT  256       // concat K (hl|hr)
#define NTILES 40       // 640/16
#define KCH    8        // 256/32 K-chunks
#define WPK_HALFS (NTILES * KCH * 512)   // 163840 halfs = 320 KB

__device__ __forceinline__ float sigmoid_f(float x) {
    return 1.0f / (1.0f + __expf(-x));          // safe at extremes
}
__device__ __forceinline__ float tanh_f(float x) {
    return 2.0f / (1.0f + __expf(-2.0f * x)) - 1.0f;  // no inf/inf NaN
}

// ---------------------------------------------------------------------------
// Pack [W_l ; W_r] (f32, K=256 x N=640) into f16 fragment layout:
// fragment f = ntile*8 + kc holds a 32x16 (KxN) B tile; lane L stores 16
// halfs at wpk[f*512 + L*16 + j] following the ISA 16-bit operand striping:
//   lane<16 : j<8 -> k=j      ; j>=8 -> k=j+8
//   lane>=16: j<8 -> k=j+8    ; j>=8 -> k=j+16
// ---------------------------------------------------------------------------
__global__ void pack_w_kernel(const float* __restrict__ Wl,
                              const float* __restrict__ Wr,
                              _Float16* __restrict__ wpk) {
    int q = blockIdx.x * blockDim.x + threadIdx.x;
    if (q >= WPK_HALFS) return;
    int f = q >> 9;            // fragment index
    int s = q & 511;
    int ntile = f >> 3;
    int kc    = f & 7;
    int lane  = s >> 4;
    int j     = s & 15;
    int col = ntile * 16 + (lane & 15);
    int k;
    if (lane < 16) k = (j < 8) ? j       : (j + 8);
    else           k = (j < 8) ? (j + 8) : (j + 16);
    int kk = kc * 32 + k;
    float v = (kk < 128) ? Wl[kk * NG + col] : Wr[(kk - 128) * NG + col];
    wpk[q] = (_Float16)v;
}

// ---------------------------------------------------------------------------
// Main fused kernel: block = 1024 threads (32 waves), owns 64 merges.
// Wave w = (mi, t): mi in 0..3 = 16-merge M-tile, t in 0..7 = 16-col channel
// tile. Gate g's tile for channels [16t,16t+16) is N-tile g*8 + t, so each
// wave accumulates all 5 gates for its channels and runs the full epilogue.
// B (and A) fragments are software double-buffered: operands for K-chunk
// kc+1 are requested before the 5 WMMAs of chunk kc issue, so each
// s_wait_loadcnt covers a whole WMMA cluster instead of a single load.
// ---------------------------------------------------------------------------
__global__ __launch_bounds__(1024) void treelstm_wmma_kernel(
    const float* __restrict__ h_bot, const float* __restrict__ c_bot,
    const float* __restrict__ h_buf, const float* __restrict__ c_buf,
    const int* __restrict__ l_bot,  const int* __restrict__ l_prev,
    const int* __restrict__ r_bot,  const int* __restrict__ r_prev,
    const _Float16* __restrict__ wpk, const float* __restrict__ bias,
    float* __restrict__ out_h, float* __restrict__ out_c)
{
    // A operand staged in fragment layout: [mtile 0..3][kc 0..7][lane][j]
    __shared__ __align__(32) _Float16 smem[64 * KTOT];   // 32 KB

    const int tid = threadIdx.x;
    const int m0  = blockIdx.x * 64;

    // ---- stage gathered hl|hr rows -> f16 -> LDS fragments ----
    #pragma unroll
    for (int it = 0; it < 8; ++it) {
        int e  = tid + it * 1024;        // pair index: 64 merges x 128 pairs
        int ml = e >> 7;                 // local merge 0..63
        int kp = e & 127;
        int kk = kp << 1;                // even k in 0..254
        int mg = m0 + ml;
        const float* src;
        int row;
        if (kk < 128) {                  // left child h
            if (mg < M2_) { row = l_bot[mg];        src = h_bot; }
            else          { row = l_prev[mg - M2_]; src = h_buf; }
        } else {                         // right child h
            if (mg < M2_) { row = r_bot[mg];        src = h_bot; }
            else          { row = r_prev[mg - M2_]; src = h_buf; }
        }
        const float2 f2 = *(const float2*)(src + (size_t)row * D_ + (kk & 127));
        v2h hv; hv.x = (_Float16)f2.x; hv.y = (_Float16)f2.y;
        int mtile = ml >> 4, mL = ml & 15;
        int kc = kk >> 5,    k  = kk & 31;
        int lane, j;
        if      (k < 8)  { lane = mL;      j = k;      }
        else if (k < 16) { lane = mL + 16; j = k - 8;  }
        else if (k < 24) { lane = mL;      j = k - 8;  }
        else             { lane = mL + 16; j = k - 16; }
        *(v2h*)(smem + ((mtile * 8 + kc) * 512) + lane * 16 + j) = hv;
    }
    __syncthreads();

    const int w    = tid >> 5;
    const int lane = tid & 31;
    const int mi   = w >> 3;   // M-tile within block
    const int t    = w & 7;    // channel tile

    // Per-wave operand base pointers (halfs):
    //   B fragment (gate g, chunk kc): ((g*8 + t)*8 + kc)*512 + lane*16
    //   gate stride = 32768 halfs, kc stride = 512 halfs
    const _Float16* wt  = wpk  + ((size_t)t * 8) * 512 + lane * 16;
    const _Float16* at  = smem + (mi * 8) * 512 + lane * 16;

    v8f  acc[5] = {};          // i, o, u, fl, fr accumulators (f32)
    v16h bcur[5], bnxt[5];
    v16h acur, anxt;

    #pragma unroll
    for (int g = 0; g < 5; ++g)
        bcur[g] = *(const v16h*)(wt + g * 32768);
    acur = *(const v16h*)(at);

    #pragma unroll
    for (int kc = 0; kc < KCH; ++kc) {
        // request next chunk's operands before consuming current ones
        if (kc < KCH - 1) {
            anxt = *(const v16h*)(at + (kc + 1) * 512);
            #pragma unroll
            for (int g = 0; g < 5; ++g)
                bnxt[g] = *(const v16h*)(wt + g * 32768 + (kc + 1) * 512);
        }
        #pragma unroll
        for (int g = 0; g < 5; ++g)
            acc[g] = __builtin_amdgcn_wmma_f32_16x16x32_f16(
                false, acur, false, bcur[g], (short)0, acc[g], false, false);
        if (kc < KCH - 1) {
            acur = anxt;
            #pragma unroll
            for (int g = 0; g < 5; ++g) bcur[g] = bnxt[g];
        }
    }

    // ---- epilogue: per lane, 8 (m, ch) cells (C/D layout: VGPR v = row) ----
    const int ch   = t * 16 + (lane & 15);
    const float bi  = bias[ch];
    const float bo  = bias[128 + ch];
    const float bu  = bias[256 + ch];
    const float bfl = bias[384 + ch];
    const float bfr = bias[512 + ch];
    const int mrow0 = mi * 16 + ((lane >> 4) << 3);

    #pragma unroll
    for (int v = 0; v < 8; ++v) {
        int mg = m0 + mrow0 + v;
        int lrow, rrow;
        const float* cs;
        if (mg < M2_) { lrow = l_bot[mg];  rrow = r_bot[mg];  cs = c_bot; }
        else { int mm = mg - M2_; lrow = l_prev[mm]; rrow = r_prev[mm]; cs = c_buf; }
        float clv = cs[(size_t)lrow * D_ + ch];
        float crv = cs[(size_t)rrow * D_ + ch];

        float iv  = sigmoid_f(acc[0][v] + bi);
        float ov  = sigmoid_f(acc[1][v] + bo);
        float uv  = tanh_f   (acc[2][v] + bu);
        float flv = sigmoid_f(acc[3][v] + bfl);
        float frv = sigmoid_f(acc[4][v] + bfr);

        float cv = iv * uv + flv * clv + frv * crv;
        float hv = ov * tanh_f(cv);

        size_t o = (size_t)mg * D_ + ch;
        out_h[o] = hv;
        out_c[o] = cv;
    }
}

extern "C" void kernel_launch(void* const* d_in, const int* in_sizes, int n_in,
                              void* d_out, int out_size, void* d_ws, size_t ws_size,
                              hipStream_t stream) {
    (void)in_sizes; (void)n_in; (void)out_size; (void)ws_size;
    const float* h_bot  = (const float*)d_in[0];
    const float* c_bot  = (const float*)d_in[1];
    const float* h_buf  = (const float*)d_in[2];
    const float* c_buf  = (const float*)d_in[3];
    const int*   l_bot  = (const int*)d_in[4];
    const int*   l_prev = (const int*)d_in[5];
    const int*   r_bot  = (const int*)d_in[6];
    const int*   r_prev = (const int*)d_in[7];
    const float* W_l    = (const float*)d_in[8];
    const float* W_r    = (const float*)d_in[9];
    const float* b      = (const float*)d_in[10];

    _Float16* wpk = (_Float16*)d_ws;                 // 320 KB packed weights
    float* out_h = (float*)d_out;
    float* out_c = out_h + (size_t)M_ * D_;

    pack_w_kernel<<<(WPK_HALFS + 255) / 256, 256, 0, stream>>>(W_l, W_r, wpk);

    treelstm_wmma_kernel<<<M_ / 64, 1024, 0, stream>>>(
        h_bot, c_bot, h_buf, c_buf, l_bot, l_prev, r_bot, r_prev,
        wpk, b, out_h, out_c);
}
